// R_NEM_45157286150897
// MI455X (gfx1250) — matrix operand order repaired and checked by reading, not compiled, not run on hardware
//
#include <hip/hip_runtime.h>
#include <hip/hip_bf16.h>

// ---------------------------------------------------------------------------
// Types for CDNA5 WMMA
// ---------------------------------------------------------------------------
typedef __bf16 v16bf __attribute__((ext_vector_type(16)));
typedef float  v8f   __attribute__((ext_vector_type(8)));

// ---------------------------------------------------------------------------
// bf16 helpers (round-to-nearest-even)
// ---------------------------------------------------------------------------
__device__ __forceinline__ unsigned short f2bf(float f) {
    unsigned int u = __float_as_uint(f);
    u += 0x7FFFu + ((u >> 16) & 1u);
    return (unsigned short)(u >> 16);
}
__device__ __forceinline__ float bf2f(unsigned short h) {
    return __uint_as_float(((unsigned int)h) << 16);
}

// ---------------------------------------------------------------------------
// Model dims
// ---------------------------------------------------------------------------
#define KK      8
#define B_ENT   4096
#define MM      4096
#define H_IN    512
#define H_ENC   256
#define H_CORE  512
#define H_CTX   256
#define H_STATE 512
#define ROWS_CORE (B_ENT * (KK - 1))   // 28672

// ---------------------------------------------------------------------------
// fp32 -> bf16 convert, 4 elements / thread
// ---------------------------------------------------------------------------
__global__ __launch_bounds__(256) void k_cvt4(const float* __restrict__ s,
                                              unsigned short* __restrict__ d, int n4) {
    int i = blockIdx.x * 256 + threadIdx.x;
    if (i >= n4) return;
    float4 f = ((const float4*)s)[i];
    uint2 o;
    o.x = (unsigned)f2bf(f.x) | ((unsigned)f2bf(f.y) << 16);
    o.y = (unsigned)f2bf(f.z) | ((unsigned)f2bf(f.w) << 16);
    ((uint2*)d)[i] = o;
}

// ---------------------------------------------------------------------------
// W [K,N] fp32 row-major  ->  Wt [N,K] bf16 (B^T, rows contiguous in K)
// ---------------------------------------------------------------------------
__global__ __launch_bounds__(256) void k_transpose_bf16(const float* __restrict__ W,
                                                        unsigned short* __restrict__ Wt,
                                                        int K, int N) {
    int i = blockIdx.x * 256 + threadIdx.x;
    if (i >= K * N) return;
    int k = i / N, n = i - k * N;
    Wt[(long)n * K + k] = f2bf(W[i]);
}

// ---------------------------------------------------------------------------
// Generic bf16 GEMM, f32 accumulate via V_WMMA_F32_16X16X32_BF16.
//   A  : M x K bf16 row-major (lda = K)
//   Bt : N x K bf16 row-major (pre-transposed weights, ldb = K)
//   C  : M x N; fp32 out (Cf) and/or bf16 out (Cb), row stride ldc
// Block = 256 threads = 8 waves; wave w -> M rows [by*128 + w*16, +16),
// N cols [bx*64, +64) as 4 WMMA tiles.  B tile staged through LDS.
// EPI: 0=none 1=relu 2=elu 3=sigmoid
// ---------------------------------------------------------------------------
template <int EPI>
__global__ __launch_bounds__(256) void k_gemm(const unsigned short* __restrict__ A, int lda,
                                              const unsigned short* __restrict__ Bt, int ldb,
                                              const float* __restrict__ bias,
                                              float* __restrict__ Cf,
                                              unsigned short* __restrict__ Cb, int ldc,
                                              int K) {
    __shared__ unsigned short ldsB[64 * 40];  // 64 cols x 32 k, padded to 40 (no bank conflicts)

    const int tid   = threadIdx.x;
    const int wave  = tid >> 5;
    const int lane  = tid & 31;
    const int lrow  = lane & 15;
    const int khalf = lane >> 4;

    const long mbase = (long)blockIdx.y * 128 + wave * 16;
    const long nbase = (long)blockIdx.x * 64;

    v8f acc[4] = {};

    // A fragment pointer: lane holds row (mbase+lrow); khalf selects the K sub-strip
    const unsigned short* aptr = A + (mbase + lrow) * (long)lda + khalf * 8;

    // cooperative B-tile staging indices (256 threads move 64 rows x 32 bf16)
    const int bn = tid >> 2;
    const int bc = (tid & 3) * 8;
    const unsigned short* bptr = Bt + (nbase + bn) * (long)ldb + bc;
    unsigned short* ldsw = &ldsB[bn * 40 + bc];

    for (int k = 0; k < K; k += 32) {
        uint4 breg = *(const uint4*)(bptr + k);
        __builtin_prefetch(aptr + k + 128, 0, 0);   // global_prefetch_b8: next A strip

        __syncthreads();                 // previous iteration's LDS reads done
        *(uint4*)ldsw = breg;

        // A fragment: lanes 0-15 get K {0..7,16..23}, lanes 16-31 get K {8..15,24..31}
        union { uint4 u[2]; v16bf v; } af;
        af.u[0] = *(const uint4*)(aptr + k);
        af.u[1] = *(const uint4*)(aptr + k + 16);

        __syncthreads();                 // B tile visible

#pragma unroll
        for (int t = 0; t < 4; ++t) {
            // B fragment: lane col = t*16+lrow, lanes 0-15 get K {0..7,16..23}, lanes 16-31 get K {8..15,24..31}
            union { uint4 u[2]; v16bf v; } bfr;
            const unsigned short* lb = &ldsB[(t * 16 + lrow) * 40 + khalf * 8];
            bfr.u[0] = *(const uint4*)(lb);
            bfr.u[1] = *(const uint4*)(lb + 16);
            acc[t] = __builtin_amdgcn_wmma_f32_16x16x32_bf16(
                false, af.v, false, bfr.v, (short)0, acc[t], false, false);
        }
    }

    // Epilogue: C/D layout -> VGPR r holds (M = mbase + r + 8*khalf, N = nbase + t*16 + lrow)
#pragma unroll
    for (int t = 0; t < 4; ++t) {
        const long col = nbase + t * 16 + lrow;
        const float b = bias[col];
#pragma unroll
        for (int r = 0; r < 8; ++r) {
            const long row = mbase + r + 8 * khalf;
            float v = acc[t][r] + b;
            if constexpr (EPI == 1) v = fmaxf(v, 0.0f);
            else if constexpr (EPI == 2) v = (v > 0.0f) ? v : (__expf(v) - 1.0f);
            else if constexpr (EPI == 3) v = 1.0f / (1.0f + __expf(-v));
            if (Cf) Cf[row * ldc + col] = v;
            if (Cb) Cb[row * ldc + col] = f2bf(v);
        }
    }
}

// ---------------------------------------------------------------------------
// Build core_in[28672, 512] = [state1[pair], state1[self]] from total[:, 0:256]
// 128 threads/row, 4 bf16 (8B) per thread.
// ---------------------------------------------------------------------------
__global__ __launch_bounds__(256) void k_gather(const unsigned short* __restrict__ total,
                                                unsigned short* __restrict__ core_in) {
    int gid = blockIdx.x * 256 + threadIdx.x;
    int r = gid >> 7, q = gid & 127;
    if (r >= ROWS_CORE) return;
    int bi = r / 56, rem = r - bi * 56;
    int i = rem / 7, j = rem - i * 7;
    int o = j + (j >= i);
    int col = q * 4;
    int ent, scol;
    if (col < H_ENC) { ent = bi * 8 + o; scol = col; }          // cs: pair state
    else             { ent = bi * 8 + i; scol = col - H_ENC; }  // fs: self state
    uint2 v = *(const uint2*)(total + (long)ent * 1024 + scol);
    *(uint2*)(core_in + (long)r * 512 + col) = v;
}

// ---------------------------------------------------------------------------
// attention[r] = sigmoid(core_out[r,:] . W_att + b_att)   (wave per row)
// ---------------------------------------------------------------------------
__global__ __launch_bounds__(256) void k_attention(const unsigned short* __restrict__ core_out,
                                                   const float* __restrict__ watt,
                                                   const float* __restrict__ batt,
                                                   float* __restrict__ att) {
    int wave = threadIdx.x >> 5, lane = threadIdx.x & 31;
    int r = blockIdx.x * 8 + wave;
    if (r >= ROWS_CORE) return;
    const unsigned short* src = core_out + (long)r * H_CORE + lane * 16;
    const float* w = watt + lane * 16;
    float sum = 0.0f;
#pragma unroll
    for (int e = 0; e < 16; ++e) sum += bf2f(src[e]) * w[e];
#pragma unroll
    for (int off = 16; off; off >>= 1) sum += __shfl_xor(sum, off, 32);
    if (lane == 0) att[r] = 1.0f / (1.0f + __expf(-(sum + batt[0])));
}

// ---------------------------------------------------------------------------
// effectrsum: total[:,256:512] = sum_j att * context   (thread per (entity,col))
// ---------------------------------------------------------------------------
__global__ __launch_bounds__(256) void k_effectsum(const float* __restrict__ att,
                                                   const float* __restrict__ ctx,
                                                   unsigned short* __restrict__ total) {
    int gid = blockIdx.x * 256 + threadIdx.x;
    int e = gid >> 8, c = gid & 255;
    if (e >= B_ENT) return;
    int bi = e >> 3, i = e & 7;
    int rbase = bi * 56 + i * 7;
    float s = 0.0f;
#pragma unroll
    for (int j = 0; j < 7; ++j)
        s += att[rbase + j] * ctx[(long)(rbase + j) * H_CTX + c];
    total[(long)e * 1024 + H_ENC + c] = f2bf(s);
}

// ---------------------------------------------------------------------------
// Launch
// ---------------------------------------------------------------------------
extern "C" void kernel_launch(void* const* d_in, const int* in_sizes, int n_in,
                              void* d_out, int out_size, void* d_ws, size_t ws_size,
                              hipStream_t stream) {
    (void)in_sizes; (void)n_in; (void)out_size; (void)ws_size;
    const float* inputs = (const float*)d_in[0];
    const float* state  = (const float*)d_in[1];
    const float* W_in   = (const float*)d_in[2];
    const float* b_in   = (const float*)d_in[3];
    const float* W_enc  = (const float*)d_in[4];
    const float* b_enc  = (const float*)d_in[5];
    const float* W_core = (const float*)d_in[6];
    const float* b_core = (const float*)d_in[7];
    const float* W_ctx  = (const float*)d_in[8];
    const float* b_ctx  = (const float*)d_in[9];
    const float* W_att  = (const float*)d_in[10];
    const float* b_att  = (const float*)d_in[11];
    const float* W_rec  = (const float*)d_in[12];
    const float* b_rec  = (const float*)d_in[13];
    const float* W_out  = (const float*)d_in[14];
    const float* b_out  = (const float*)d_in[15];

    // workspace carve-up (256B aligned)
    char* ws = (char*)d_ws;
    size_t off = 0;
    auto alloc = [&](size_t bytes) -> char* {
        char* p = ws + off;
        off = (off + bytes + 255) & ~(size_t)255;
        return p;
    };
    unsigned short* inputs_bf = (unsigned short*)alloc((size_t)B_ENT * MM * 2);     // 32MB, reused as core_out
    unsigned short* state_bf  = (unsigned short*)alloc((size_t)B_ENT * H_STATE * 2);
    unsigned short* WinT      = (unsigned short*)alloc((size_t)H_IN * MM * 2);
    unsigned short* WencT     = (unsigned short*)alloc((size_t)H_ENC * H_STATE * 2);
    unsigned short* WcoreT    = (unsigned short*)alloc((size_t)H_CORE * 2 * H_ENC * 2);
    unsigned short* WctxT     = (unsigned short*)alloc((size_t)H_CTX * H_CORE * 2);
    unsigned short* WrecT     = (unsigned short*)alloc((size_t)H_STATE * 1024 * 2);
    unsigned short* WoutT     = (unsigned short*)alloc((size_t)MM * H_STATE * 2);
    unsigned short* total_bf  = (unsigned short*)alloc((size_t)B_ENT * 1024 * 2);
    unsigned short* core_in   = (unsigned short*)alloc((size_t)ROWS_CORE * 512 * 2);
    float*          context   = (float*)alloc((size_t)ROWS_CORE * H_CTX * 4);
    float*          att       = (float*)alloc((size_t)ROWS_CORE * 4);
    unsigned short* nstate_bf = (unsigned short*)alloc((size_t)B_ENT * H_STATE * 2);
    unsigned short* core_out  = inputs_bf;  // alias: inputs_bf dead after GEMM1

    float* out_f    = (float*)d_out;              // [4096, 4096]
    float* nstate_f = out_f + (size_t)B_ENT * MM; // [4096, 512]

    dim3 blk(256);

    // fp32 -> bf16 activations
    k_cvt4<<<(B_ENT * MM / 4 + 255) / 256, blk, 0, stream>>>(inputs, inputs_bf, B_ENT * MM / 4);
    k_cvt4<<<(B_ENT * H_STATE / 4 + 255) / 256, blk, 0, stream>>>(state, state_bf, B_ENT * H_STATE / 4);

    // weight transposes (K,N) -> bf16 (N,K)
    k_transpose_bf16<<<(MM * H_IN + 255) / 256, blk, 0, stream>>>(W_in, WinT, MM, H_IN);
    k_transpose_bf16<<<(H_STATE * H_ENC + 255) / 256, blk, 0, stream>>>(W_enc, WencT, H_STATE, H_ENC);
    k_transpose_bf16<<<(2 * H_ENC * H_CORE + 255) / 256, blk, 0, stream>>>(W_core, WcoreT, 2 * H_ENC, H_CORE);
    k_transpose_bf16<<<(H_CORE * H_CTX + 255) / 256, blk, 0, stream>>>(W_ctx, WctxT, H_CORE, H_CTX);
    k_transpose_bf16<<<(1024 * H_STATE + 255) / 256, blk, 0, stream>>>(W_rec, WrecT, 1024, H_STATE);
    k_transpose_bf16<<<(H_STATE * MM + 255) / 256, blk, 0, stream>>>(W_out, WoutT, H_STATE, MM);

    // x = elu(inputs @ W_in) -> total[:, 512:1024] (bf16)
    k_gemm<2><<<dim3(H_IN / 64, B_ENT / 128), blk, 0, stream>>>(
        inputs_bf, MM, WinT, MM, b_in, nullptr, total_bf + 512, 1024, MM);

    // state1 = relu(state @ W_enc) -> total[:, 0:256] (bf16)
    k_gemm<1><<<dim3(H_ENC / 64, B_ENT / 128), blk, 0, stream>>>(
        state_bf, H_STATE, WencT, H_STATE, b_enc, nullptr, total_bf, 1024, H_STATE);

    // core_in = [state1[pair], state1[self]]
    k_gather<<<ROWS_CORE * 128 / 256, blk, 0, stream>>>(total_bf, core_in);

    // core_out = relu(core_in @ W_core)  (bf16)
    k_gemm<1><<<dim3(H_CORE / 64, ROWS_CORE / 128), blk, 0, stream>>>(
        core_in, 512, WcoreT, 512, b_core, nullptr, core_out, H_CORE, 512);

    // context = relu(core_out @ W_ctx)  (fp32)
    k_gemm<1><<<dim3(H_CTX / 64, ROWS_CORE / 128), blk, 0, stream>>>(
        core_out, H_CORE, WctxT, H_CORE, b_ctx, context, nullptr, H_CTX, H_CORE);

    // attention = sigmoid(core_out @ W_att)
    k_attention<<<ROWS_CORE / 8, blk, 0, stream>>>(core_out, W_att, b_att, att);

    // effectrsum -> total[:, 256:512] (bf16)
    k_effectsum<<<B_ENT * 256 / 256, blk, 0, stream>>>(att, context, total_bf);

    // new_state = sigmoid(total @ W_rec) -> fp32 d_out tail + bf16 for final GEMM
    k_gemm<3><<<dim3(H_STATE / 64, B_ENT / 128), blk, 0, stream>>>(
        total_bf, 1024, WrecT, 1024, b_rec, nstate_f, nstate_bf, H_STATE, 1024);

    // out = sigmoid(new_state @ W_out) -> fp32 d_out head
    k_gemm<3><<<dim3(MM / 64, B_ENT / 128), blk, 0, stream>>>(
        nstate_bf, H_STATE, WoutT, H_STATE, b_out, out_f, nullptr, MM, H_STATE);
}
